// MABlock_40630390621012
// MI455X (gfx1250) — matrix operand order repaired and checked by reading, not compiled, not run on hardware
//
#include <hip/hip_runtime.h>

// ---------------------------------------------------------------------------
// MABlock for MI455X (gfx1250): bf16 WMMA everywhere, f32 accumulate.
// Retrieval GEMM stages K-memory chunks into LDS with gfx1250 async-to-LDS
// copies (ASYNCcnt double buffering), shared by all 8 waves of the block.
// ---------------------------------------------------------------------------

typedef __attribute__((ext_vector_type(16))) __bf16 v16bf;
typedef __attribute__((ext_vector_type(8)))  __bf16 v8bf;
typedef __attribute__((ext_vector_type(8)))  float  v8f;

#define B_    4
#define N_    512
#define BN_   2048      // B*N
#define D_    1024
#define H_    16
#define HD_   64
#define MEM_  32768
#define IPQ_  8
#define NEG_INF (-3.0e38f)

#define WMMA_BF16(a, b, c) \
  __builtin_amdgcn_wmma_f32_16x16x32_bf16(false, (a), false, (b), (short)0, (c), false, false)

// ---------------- fp32 <-> bf16 (round-to-nearest-even) --------------------
__device__ __forceinline__ __bf16 f2bf(float f) {
  unsigned u = __builtin_bit_cast(unsigned, f);
  unsigned r = u + 0x7fffu + ((u >> 16) & 1u);
  unsigned short h = (unsigned short)(r >> 16);
  return __builtin_bit_cast(__bf16, h);
}

// ---------------- WMMA fragment loaders (this file's contiguous-K order) --------------------
// A (16x32 bf16): lane L<16 holds row L, K=0..15; lane L+16: K=16..31
__device__ __forceinline__ v16bf load_a16x32(const __bf16* base, int ld, int k0) {
  int lane = threadIdx.x & 31;
  int row  = lane & 15;
  int hi   = lane >> 4;
  const __bf16* p = base + (size_t)row * ld + k0 + hi * 16;
  v8bf lo = *(const v8bf*)p;
  v8bf hh = *(const v8bf*)(p + 8);
  v16bf a;
#pragma unroll
  for (int e = 0; e < 8; ++e) { a[e] = lo[e]; a[e + 8] = hh[e]; }
  return a;
}

// B (32x16 bf16), B[k][n] = base[n*ld + k] (contiguous along K in memory):
// lanes 0-15 hold col n=lane, K=0..15; lanes 16-31 hold K=16..31
__device__ __forceinline__ v16bf load_b32x16(const __bf16* base, int ld, int k0) {
  int lane = threadIdx.x & 31;
  int c    = lane & 15;
  int hi   = lane >> 4;
  const __bf16* p = base + (size_t)c * ld + k0 + hi * 16;
  v8bf lo = *(const v8bf*)p;
  v8bf hh = *(const v8bf*)(p + 8);
  v16bf b;
#pragma unroll
  for (int e = 0; e < 8; ++e) { b[e] = lo[e]; b[e + 8] = hh[e]; }
  return b;
}

// B (32x16) where B[k][n] = base[k*ld + n]  (K-major gather, used for P*V)
__device__ __forceinline__ v16bf load_b_strided(const __bf16* base, int ld) {
  int lane = threadIdx.x & 31;
  int c    = lane & 15;
  int hi   = lane >> 4;
  const __bf16* p = base + (size_t)hi * 16 * ld + c;
  v16bf b;
#pragma unroll
  for (int e = 0; e < 16; ++e) b[e] = p[(size_t)e * ld];
  return b;
}

// -------- cross-lane reductions within the 16-lane C-layout groups --------
__device__ __forceinline__ float redmax16(float v) {
#pragma unroll
  for (int m = 1; m < 16; m <<= 1) v = fmaxf(v, __shfl_xor(v, m, 32));
  return v;
}
__device__ __forceinline__ float redsum16(float v) {
#pragma unroll
  for (int m = 1; m < 16; m <<= 1) v += __shfl_xor(v, m, 32);
  return v;
}

// -------- gfx1250 async global->LDS copy (tracked by ASYNCcnt) ------------
__device__ __forceinline__ void async_to_lds_b128(unsigned ldsOff, const __bf16* g) {
  unsigned long long ga = (unsigned long long)(uintptr_t)g;
  asm volatile("global_load_async_to_lds_b128 %0, %1, off"
               :: "v"(ldsOff), "v"(ga) : "memory");
}
__device__ __forceinline__ void wait_async_le2() {
  asm volatile("s_wait_asynccnt 0x2" ::: "memory");
}
__device__ __forceinline__ void wait_async_0() {
  asm volatile("s_wait_asynccnt 0x0" ::: "memory");
}

// ---------------------------------------------------------------------------
// K0: f32 -> bf16 convert
// ---------------------------------------------------------------------------
__global__ void cvt_bf16(const float* __restrict__ s, __bf16* __restrict__ d, int n) {
  for (int i = blockIdx.x * blockDim.x + threadIdx.x; i < n; i += gridDim.x * blockDim.x)
    d[i] = f2bf(s[i]);
}

// ---------------------------------------------------------------------------
// K1: Y[M x Nc] = A[M x K](bf16) @ W[Nc x K]^T (bf16), f32 accum.
// block = 128 threads (4 waves); block tile 64 rows x 256 cols; wave tile 64x64.
// mode 0: Ybf = bf16(scale*acc)   mode 1: Yf = sigmoid(acc)   mode 2: Yf = acc
// ---------------------------------------------------------------------------
__global__ __launch_bounds__(128) void proj_wmma(
    const __bf16* __restrict__ A, const __bf16* __restrict__ W,
    __bf16* __restrict__ Ybf, float* __restrict__ Yf,
    int Ncols, int K, int mode, float scale) {
  int wid  = threadIdx.x >> 5;
  int lane = threadIdx.x & 31;
  int m0   = blockIdx.y * 64;
  int n0   = blockIdx.x * 256 + wid * 64;

  v8f acc[4][4] = {};
  for (int k0 = 0; k0 < K; k0 += 32) {
    v16bf af[4];
#pragma unroll
    for (int mt = 0; mt < 4; ++mt)
      af[mt] = load_a16x32(A + (size_t)(m0 + mt * 16) * K, K, k0);
#pragma unroll
    for (int nt = 0; nt < 4; ++nt) {
      v16bf bf = load_b32x16(W + (size_t)(n0 + nt * 16) * K, K, k0);
#pragma unroll
      for (int mt = 0; mt < 4; ++mt)
        acc[mt][nt] = WMMA_BF16(af[mt], bf, acc[mt][nt]);
    }
  }
  int n = lane & 15, mh = (lane >> 4) * 8;
#pragma unroll
  for (int mt = 0; mt < 4; ++mt)
#pragma unroll
    for (int nt = 0; nt < 4; ++nt)
#pragma unroll
      for (int r = 0; r < 8; ++r) {
        size_t o = (size_t)(m0 + mt * 16 + mh + r) * Ncols + (n0 + nt * 16 + n);
        float v = acc[mt][nt][r] * scale;
        if (mode == 0)       Ybf[o] = f2bf(v);
        else if (mode == 1)  Yf[o]  = 1.0f / (1.0f + __expf(-v));
        else                 Yf[o]  = v;
      }
}

// ---------------------------------------------------------------------------
// K2: FIFO memory build, head-major bf16: kmem[h][r][d]
// ---------------------------------------------------------------------------
__global__ void build_mem(const float* __restrict__ km0, const float* __restrict__ vm0,
                          const __bf16* __restrict__ kb, const __bf16* __restrict__ vb,
                          __bf16* __restrict__ kmem, __bf16* __restrict__ vmem) {
  size_t t = (size_t)blockIdx.x * blockDim.x + threadIdx.x;
  int d = (int)(t & 63);
  int r = (int)((t >> 6) & (MEM_ - 1));
  int h = (int)(t >> 21);
  if (r < MEM_ - BN_) {
    size_t s = ((size_t)h * MEM_ + r + BN_) * HD_ + d;
    kmem[t] = f2bf(km0[s]);
    vmem[t] = f2bf(vm0[s]);
  } else {
    int i = r - (MEM_ - BN_);
    size_t s = (size_t)i * D_ + h * HD_ + d;
    kmem[t] = kb[s];
    vmem[t] = vb[s];
  }
}

// ---------------------------------------------------------------------------
// K3: fused retrieval sims + top-8.
// Block = 8 waves sharing one head; 64x64 key chunks are double-buffered in
// LDS via global_load_async_to_lds_b128 (2 issues/wave/chunk, ASYNCcnt).
// Each wave owns 32 queries; lane l keeps a register top-8 for query q0+l.
// ---------------------------------------------------------------------------
__global__ __launch_bounds__(256) void retrieval_topk(
    const __bf16* __restrict__ qb, const __bf16* __restrict__ kmem,
    int* __restrict__ idxout) {
  __shared__ float sc[8][32][64];                       // 64 KB
  __shared__ __align__(16) __bf16 kstage[2][64][64];    // 2 x 8 KB ping-pong
  int tid  = threadIdx.x;
  int wid  = tid >> 5;
  int lane = tid & 31;
  int h    = blockIdx.x >> 3;                           // 16 heads
  int q0   = (blockIdx.x & 7) * 256 + wid * 32;         // 256 queries / block

  v16bf aq[2][2];
#pragma unroll
  for (int mt = 0; mt < 2; ++mt)
#pragma unroll
    for (int kt = 0; kt < 2; ++kt)
      aq[mt][kt] = load_a16x32(qb + (size_t)(q0 + mt * 16) * D_ + h * HD_, D_, kt * 32);

  float tv[8]; int ti[8];
#pragma unroll
  for (int t = 0; t < 8; ++t) { tv[t] = NEG_INF; ti[t] = 0; }
  float vmin = NEG_INF; int imin = 0;

  const __bf16* kh = kmem + (size_t)h * MEM_ * HD_;
  unsigned ldsBase = (unsigned)(uintptr_t)&kstage[0][0][0];
  int n = lane & 15, mh = (lane >> 4) * 8;

  // stage chunk 0 into buffer 0 (each thread copies 2 x 16B of the 8KB chunk)
  {
    const __bf16* g = kh + (size_t)tid * 8;
    async_to_lds_b128(ldsBase + tid * 16u,         g);
    async_to_lds_b128(ldsBase + tid * 16u + 4096u, g + 2048);
  }

  for (int c0 = 0; c0 < MEM_; c0 += 64) {
    int cur = (c0 >> 6) & 1;
    if (c0 + 64 < MEM_) {
      // buffer cur^1 was last read before the previous iteration's barrier
      const __bf16* g = kh + (size_t)(c0 + 64) * HD_ + (size_t)tid * 8;
      unsigned dst = ldsBase + (unsigned)(cur ^ 1) * 8192u + tid * 16u;
      async_to_lds_b128(dst,         g);
      async_to_lds_b128(dst + 4096u, g + 2048);
      wait_async_le2();   // the 2 newest remain in flight; chunk `cur` landed
    } else {
      wait_async_0();     // tail: nothing new issued, drain everything
    }
    __syncthreads();      // make all waves' staged bytes visible

    const __bf16* kbuf = &kstage[cur][0][0];
    v8f acc[2][4] = {};
#pragma unroll
    for (int nt = 0; nt < 4; ++nt)
#pragma unroll
      for (int kt = 0; kt < 2; ++kt) {
        v16bf bf = load_b32x16(kbuf + (size_t)(nt * 16) * HD_, HD_, kt * 32);
        acc[0][nt] = WMMA_BF16(aq[0][kt], bf, acc[0][nt]);
        acc[1][nt] = WMMA_BF16(aq[1][kt], bf, acc[1][nt]);
      }
#pragma unroll
    for (int mt = 0; mt < 2; ++mt)
#pragma unroll
      for (int nt = 0; nt < 4; ++nt)
#pragma unroll
        for (int r = 0; r < 8; ++r)
          sc[wid][mt * 16 + mh + r][nt * 16 + n] = acc[mt][nt][r];
    __syncthreads();      // also fences this chunk's kstage reads
#pragma unroll 4
    for (int j = 0; j < 64; ++j) {
      float v = sc[wid][lane][j];
      if (v > vmin) {
        tv[imin] = v; ti[imin] = c0 + j;
        vmin = tv[0]; imin = 0;
#pragma unroll
        for (int t = 1; t < 8; ++t)
          if (tv[t] < vmin) { vmin = tv[t]; imin = t; }
      }
    }
    __syncthreads();
  }
#pragma unroll
  for (int t = 0; t < 8; ++t)
    idxout[((size_t)h * BN_ + q0 + lane) * IPQ_ + t] = ti[t];
}

// ---------------------------------------------------------------------------
// K4: gather retrieved K/V into per-(b,h) dense [4096 x 64] bf16 tiles
// ---------------------------------------------------------------------------
__global__ void gather_kv(const int* __restrict__ idx,
                          const __bf16* __restrict__ kmem, const __bf16* __restrict__ vmem,
                          __bf16* __restrict__ K2, __bf16* __restrict__ V2) {
  size_t g = (size_t)blockIdx.x * blockDim.x + threadIdx.x;
  int d  = (int)(g & 63);
  int m  = (int)((g >> 6) & 4095);
  int h  = (int)((g >> 18) & 15);
  int b  = (int)(g >> 22);
  int nq = m >> 3, p = m & 7;
  int r  = idx[((size_t)h * BN_ + b * N_ + nq) * IPQ_ + p];
  size_t src = ((size_t)h * MEM_ + r) * HD_ + d;
  K2[g] = kmem[src];
  V2[g] = vmem[src];
}

// ---------------------------------------------------------------------------
// K5: online-softmax (flash) attention, one wave per (b, h, 32 queries).
// ---------------------------------------------------------------------------
__global__ __launch_bounds__(256) void attn_wmma(
    const __bf16* __restrict__ qb, const __bf16* __restrict__ kbase,
    const __bf16* __restrict__ vbase, __bf16* __restrict__ outh,
    int Mkv, int mode) {
  __shared__ __align__(16) __bf16 pbuf[8][32][32];
  int wid  = threadIdx.x >> 5;
  int lane = threadIdx.x & 31;
  int unit = blockIdx.x * 8 + wid;
  int qt   = unit & 15;
  int h    = (unit >> 4) & 15;
  int b    = unit >> 8;
  int q0   = b * N_ + qt * 32;

  const __bf16 *Kp, *Vp; int ldk;
  if (mode == 0) {
    ldk = D_;
    Kp = kbase + (size_t)b * N_ * D_ + h * HD_;
    Vp = vbase + (size_t)b * N_ * D_ + h * HD_;
  } else {
    ldk = HD_;
    size_t o = ((size_t)(b * H_ + h)) * (size_t)Mkv * HD_;
    Kp = kbase + o;
    Vp = vbase + o;
  }

  v16bf aq[2][2];
#pragma unroll
  for (int mt = 0; mt < 2; ++mt)
#pragma unroll
    for (int kt = 0; kt < 2; ++kt)
      aq[mt][kt] = load_a16x32(qb + (size_t)(q0 + mt * 16) * D_ + h * HD_, D_, kt * 32);

  v8f o[2][4] = {};
  float mst[2][8], lst[2][8];
#pragma unroll
  for (int mt = 0; mt < 2; ++mt)
#pragma unroll
    for (int r = 0; r < 8; ++r) { mst[mt][r] = NEG_INF; lst[mt][r] = 0.0f; }

  int n = lane & 15, mh = (lane >> 4) * 8;

  for (int c0 = 0; c0 < Mkv; c0 += 32) {
    v8f s[2][2] = {};
#pragma unroll
    for (int nt = 0; nt < 2; ++nt)
#pragma unroll
      for (int kt = 0; kt < 2; ++kt) {
        v16bf bf = load_b32x16(Kp + (size_t)(c0 + nt * 16) * ldk, ldk, kt * 32);
        s[0][nt] = WMMA_BF16(aq[0][kt], bf, s[0][nt]);
        s[1][nt] = WMMA_BF16(aq[1][kt], bf, s[1][nt]);
      }
#pragma unroll
    for (int mt = 0; mt < 2; ++mt)
#pragma unroll
      for (int r = 0; r < 8; ++r) {
        float mx   = redmax16(fmaxf(s[mt][0][r], s[mt][1][r]));
        float mi   = fmaxf(mst[mt][r], mx);
        float corr = __expf(mst[mt][r] - mi);
        float p0   = __expf(s[mt][0][r] - mi);
        float p1   = __expf(s[mt][1][r] - mi);
        float rs   = redsum16(p0 + p1);
        lst[mt][r] = lst[mt][r] * corr + rs;
        mst[mt][r] = mi;
        pbuf[wid][mt * 16 + mh + r][n]      = f2bf(p0);
        pbuf[wid][mt * 16 + mh + r][16 + n] = f2bf(p1);
#pragma unroll
        for (int dt = 0; dt < 4; ++dt) o[mt][dt][r] *= corr;
      }
    __syncthreads();
    v16bf pa[2];
    pa[0] = load_a16x32(&pbuf[wid][0][0],  32, 0);
    pa[1] = load_a16x32(&pbuf[wid][16][0], 32, 0);
#pragma unroll
    for (int dt = 0; dt < 4; ++dt) {
      v16bf bv = load_b_strided(Vp + (size_t)c0 * ldk + dt * 16, ldk);
      o[0][dt] = WMMA_BF16(pa[0], bv, o[0][dt]);
      o[1][dt] = WMMA_BF16(pa[1], bv, o[1][dt]);
    }
    __syncthreads();
  }
#pragma unroll
  for (int mt = 0; mt < 2; ++mt)
#pragma unroll
    for (int dt = 0; dt < 4; ++dt)
#pragma unroll
      for (int r = 0; r < 8; ++r)
        outh[(size_t)(q0 + mt * 16 + mh + r) * D_ + h * HD_ + dt * 16 + n] =
            f2bf(o[mt][dt][r] / lst[mt][r]);
}

// ---------------------------------------------------------------------------
// K6: out = w*a + (1-w)*am + bo
// ---------------------------------------------------------------------------
__global__ void blend_out(const float* __restrict__ w, const float* __restrict__ a,
                          const float* __restrict__ am, const float* __restrict__ bo,
                          float* __restrict__ out, int nTot) {
  int i = blockIdx.x * blockDim.x + threadIdx.x;
  if (i < nTot) {
    float ww = w[i];
    out[i] = ww * a[i] + (1.0f - ww) * am[i] + bo[i & (D_ - 1)];
  }
}

// ---------------------------------------------------------------------------
extern "C" void kernel_launch(void* const* d_in, const int* in_sizes, int n_in,
                              void* d_out, int out_size, void* d_ws, size_t ws_size,
                              hipStream_t stream) {
  (void)in_sizes; (void)n_in; (void)out_size; (void)ws_size;
  const float* x   = (const float*)d_in[0];
  const float* Wq  = (const float*)d_in[1];
  const float* Wk  = (const float*)d_in[2];
  const float* Wv  = (const float*)d_in[3];
  const float* Ww  = (const float*)d_in[4];
  const float* Wo  = (const float*)d_in[5];
  const float* bo  = (const float*)d_in[6];
  const float* km0 = (const float*)d_in[7];
  const float* vm0 = (const float*)d_in[8];

  char* ws = (char*)d_ws;
  size_t off = 0;
  auto alloc = [&](size_t bytes) -> void* {
    void* p = ws + off;
    off += (bytes + 255) & ~(size_t)255;
    return p;
  };

  __bf16* xb    = (__bf16*)alloc((size_t)BN_ * D_ * 2);
  __bf16* wqb   = (__bf16*)alloc((size_t)D_ * D_ * 2);
  __bf16* wkb   = (__bf16*)alloc((size_t)D_ * D_ * 2);
  __bf16* wvb   = (__bf16*)alloc((size_t)D_ * D_ * 2);
  __bf16* wwb   = (__bf16*)alloc((size_t)D_ * D_ * 2);
  __bf16* wob   = (__bf16*)alloc((size_t)D_ * D_ * 2);
  __bf16* qb    = (__bf16*)alloc((size_t)BN_ * D_ * 2);
  __bf16* kb    = (__bf16*)alloc((size_t)BN_ * D_ * 2);
  __bf16* vb    = (__bf16*)alloc((size_t)BN_ * D_ * 2);
  float*  wg    = (float*)alloc((size_t)BN_ * D_ * 4);
  __bf16* kmemb = (__bf16*)alloc((size_t)H_ * MEM_ * HD_ * 2);
  __bf16* vmemb = (__bf16*)alloc((size_t)H_ * MEM_ * HD_ * 2);
  int*    idxb  = (int*)alloc((size_t)H_ * BN_ * IPQ_ * 4);
  __bf16* K2    = (__bf16*)alloc((size_t)B_ * H_ * (N_ * IPQ_) * HD_ * 2);
  __bf16* V2    = (__bf16*)alloc((size_t)B_ * H_ * (N_ * IPQ_) * HD_ * 2);
  __bf16* ah    = (__bf16*)alloc((size_t)BN_ * D_ * 2);
  __bf16* amh   = (__bf16*)alloc((size_t)BN_ * D_ * 2);
  float*  af    = (float*)alloc((size_t)BN_ * D_ * 4);
  float*  amf   = (float*)alloc((size_t)BN_ * D_ * 4);

  const float SCALE = 0.125f;  // 1/sqrt(64)

  // ---- bf16 conversions ----
  cvt_bf16<<<8192, 256, 0, stream>>>(x,  xb,  BN_ * D_);
  cvt_bf16<<<4096, 256, 0, stream>>>(Wq, wqb, D_ * D_);
  cvt_bf16<<<4096, 256, 0, stream>>>(Wk, wkb, D_ * D_);
  cvt_bf16<<<4096, 256, 0, stream>>>(Wv, wvb, D_ * D_);
  cvt_bf16<<<4096, 256, 0, stream>>>(Ww, wwb, D_ * D_);
  cvt_bf16<<<4096, 256, 0, stream>>>(Wo, wob, D_ * D_);

  // ---- projections (q gets the attention scale folded in; top-k is
  //      scale-invariant so retrieval ordering is unchanged) ----
  dim3 pg(D_ / 256, BN_ / 64);
  proj_wmma<<<pg, 128, 0, stream>>>(xb, wqb, qb, nullptr, D_, D_, 0, SCALE);
  proj_wmma<<<pg, 128, 0, stream>>>(xb, wkb, kb, nullptr, D_, D_, 0, 1.0f);
  proj_wmma<<<pg, 128, 0, stream>>>(xb, wvb, vb, nullptr, D_, D_, 0, 1.0f);
  proj_wmma<<<pg, 128, 0, stream>>>(xb, wwb, nullptr, wg, D_, D_, 1, 1.0f);

  // ---- self attention over fresh keys ----
  attn_wmma<<<128, 256, 0, stream>>>(qb, kb, vb, ah, N_, 0);

  // ---- FIFO memory update (bf16, head-major) ----
  build_mem<<<(H_ * MEM_ * HD_) / 256, 256, 0, stream>>>(km0, vm0, kb, vb, kmemb, vmemb);

  // ---- retrieval: fused sims GEMM + top-8 (async LDS staged) ----
  retrieval_topk<<<128, 256, 0, stream>>>(qb, kmemb, idxb);

  // ---- gather retrieved K/V ----
  gather_kv<<<(B_ * H_ * (N_ * IPQ_) * HD_) / 256, 256, 0, stream>>>(idxb, kmemb, vmemb, K2, V2);

  // ---- attention over retrieved memories ----
  attn_wmma<<<128, 256, 0, stream>>>(qb, K2, V2, amh, N_ * IPQ_, 1);

  // ---- output projections ----
  proj_wmma<<<pg, 128, 0, stream>>>(ah,  wob, nullptr, af,  D_, D_, 2, 1.0f);
  proj_wmma<<<pg, 128, 0, stream>>>(amh, wob, nullptr, amf, D_, D_, 2, 1.0f);

  // ---- gated blend ----
  blend_out<<<(BN_ * D_) / 256, 256, 0, stream>>>(wg, af, amf, bo, (float*)d_out, BN_ * D_);
}
